// RNN_35347580846088
// MI455X (gfx1250) — matrix operand order repaired and checked by reading, not compiled, not run on hardware
//
#include <hip/hip_runtime.h>

#define SEQ 1024
#define BATCHN 256
#define IN_DIM 64
#define HID 128

typedef _Float16 v16h __attribute__((ext_vector_type(16)));
typedef float    v8f  __attribute__((ext_vector_type(8)));

// Branchless tanh: CDNA5 hardware transcendental (co-executes with WMMA/XDL).
__device__ __forceinline__ float fast_tanh(float x) {
#if __has_builtin(__builtin_amdgcn_tanhf)
  return __builtin_amdgcn_tanhf(x);
#else
#if __has_builtin(__builtin_amdgcn_exp2f)
  float e = __builtin_amdgcn_exp2f(x * 2.885390081777927f);
#else
  float e = __expf(2.0f * x);
#endif
#if __has_builtin(__builtin_amdgcn_rcpf)
  return 1.0f - 2.0f * __builtin_amdgcn_rcpf(e + 1.0f);
#else
  return 1.0f - 2.0f / (e + 1.0f);
#endif
#endif
}

// B operand (32x16 f16) chunk from row-major f32 weight W[n][k] (B[k][n] = W[n][k]).
// lanes 0-15 -> N=lane, K=kbase+0..15 ; lanes 16-31 -> N=lane-16, K=kbase+16..31.
__device__ __forceinline__ v16h load_B_from_f32(const float* __restrict__ W, int ld,
                                                int n, int lane, int kbase) {
  int koff = kbase + ((lane >> 4) << 4);
  const float* r = W + n * ld + koff;
  v16h b;
#pragma unroll
  for (int j = 0; j < 16; ++j) b[j] = (_Float16)r[j];
  return b;
}

// A operand (16x32 f16) chunk from LDS row-major [16][ld].
// lanes 0-15: M=lane, K = klo+0..15 ; lanes 16-31: +16 on klo.
__device__ __forceinline__ v16h load_A_lds(const _Float16* p, int lane, int chunk, int ld) {
  int m = lane & 15;
  int klo = chunk * 32 + ((lane >> 4) << 4);
  const _Float16* r = p + m * ld + klo;
  v16h a;
#pragma unroll
  for (int j = 0; j < 16; ++j) a[j] = r[j];
  return a;
}

__global__ __launch_bounds__(256) void rnn2_fused(
    const float* __restrict__ x,
    const float* __restrict__ Wih0, const float* __restrict__ bih0,
    const float* __restrict__ Whh0, const float* __restrict__ bhh0,
    const float* __restrict__ Wih1, const float* __restrict__ bih1,
    const float* __restrict__ Whh1, const float* __restrict__ bhh1,
    float* __restrict__ out)  // y1 (S,B,H) then h_n (2,B,H)
{
  __shared__ _Float16 sh_h0[2][16 * HID];   // h0(t) lives in buffer t&1
  __shared__ _Float16 sh_h1[2][16 * HID];   // h1(t) lives in buffer t&1
  __shared__ _Float16 sh_x[2][16 * IN_DIM]; // x(t) lives in buffer t&1

  const int tid   = threadIdx.x;
  const int lane  = tid & 31;
  const int wave  = tid >> 5;
  const int ncol  = wave * 16;
  const int bbase = blockIdx.x * 16;

  // ---- weights resident in VGPRs as WMMA B operands (f16) ----
  v16h B_ih0[2], B_hh0[4], B_ih1[4], B_hh1[4];
  {
    int n = ncol + (lane & 15);
#pragma unroll
    for (int c = 0; c < 2; ++c) B_ih0[c] = load_B_from_f32(Wih0, IN_DIM, n, lane, c * 32);
#pragma unroll
    for (int c = 0; c < 4; ++c) B_hh0[c] = load_B_from_f32(Whh0, HID, n, lane, c * 32);
#pragma unroll
    for (int c = 0; c < 4; ++c) B_ih1[c] = load_B_from_f32(Wih1, HID, n, lane, c * 32);
#pragma unroll
    for (int c = 0; c < 4; ++c) B_hh1[c] = load_B_from_f32(Whh1, HID, n, lane, c * 32);
  }
  const int   mycol  = ncol + (lane & 15);
  const int   rowoff = (lane >> 4) << 3;    // C/D tile: VGPR r -> row rowoff+r, col mycol
  const float b0s = bih0[mycol] + bhh0[mycol];
  const float b1s = bih1[mycol] + bhh1[mycol];

  // h1(-1) = 0 lives in buffer 1 (read at superstep t=1)
  for (int i = tid; i < 16 * HID; i += 256) sh_h1[1][i] = (_Float16)0.f;

  float* yout = out;
  float* hn   = out + (size_t)SEQ * BATCHN * HID;

  // ---- prologue: stage x(0); h0(0) = tanh(x0 W_ih0^T + b0)  (h_init = 0) ----
  {
    const float4* src = (const float4*)(x + ((size_t)0 * BATCHN + bbase) * IN_DIM);
    float4 v = src[tid];
    _Float16* dx = sh_x[0];
    dx[tid * 4 + 0] = (_Float16)v.x; dx[tid * 4 + 1] = (_Float16)v.y;
    dx[tid * 4 + 2] = (_Float16)v.z; dx[tid * 4 + 3] = (_Float16)v.w;
  }
  __syncthreads();
  {
    v8f acc;
#pragma unroll
    for (int j = 0; j < 8; ++j) acc[j] = b0s;
#pragma unroll
    for (int c = 0; c < 2; ++c) {
      v16h a = load_A_lds(sh_x[0], lane, c, IN_DIM);
      acc = __builtin_amdgcn_wmma_f32_16x16x32_f16(false, a, false, B_ih0[c],
                                                   (short)0, acc, false, false);
    }
#pragma unroll
    for (int j = 0; j < 8; ++j) acc[j] = fast_tanh(acc[j]);
#pragma unroll
    for (int r = 0; r < 8; ++r)
      sh_h0[0][(rowoff + r) * HID + mycol] = (_Float16)acc[r];
  }

  // ---- main loop: superstep t computes h0(t) and h1(t-1) behind ONE barrier ----
  for (int t = 1; t < SEQ; ++t) {
    const int pa = t & 1;       // buffer of h0(t), x(t); also buffer of h1(t-2)
    const int qa = pa ^ 1;      // buffer of h0(t-1); also buffer of h1(t-1)

    // stage x(t) -> sh_x[pa]; prefetch x(t+1)
    {
      const float4* src = (const float4*)(x + ((size_t)t * BATCHN + bbase) * IN_DIM);
      float4 v = src[tid];
      _Float16* dx = sh_x[pa];
      dx[tid * 4 + 0] = (_Float16)v.x; dx[tid * 4 + 1] = (_Float16)v.y;
      dx[tid * 4 + 2] = (_Float16)v.z; dx[tid * 4 + 3] = (_Float16)v.w;
      if (t + 1 < SEQ)
        __builtin_prefetch(x + ((size_t)(t + 1) * BATCHN + bbase) * IN_DIM + tid * 4, 0, 3);
    }
    __syncthreads();

    // chain 0: h0(t) = tanh(x(t) W_ih0^T + b0 + h0(t-1) W_hh0^T)
    v8f acc;
#pragma unroll
    for (int j = 0; j < 8; ++j) acc[j] = b0s;
    // chain 1: h1(t-1) = tanh(h0(t-1) W_ih1^T + b1 + h1(t-2) W_hh1^T)
    v8f acc1;
#pragma unroll
    for (int j = 0; j < 8; ++j) acc1[j] = b1s;

    // interleave the two independent WMMA chains
#pragma unroll
    for (int c = 0; c < 2; ++c) {
      v16h a0 = load_A_lds(sh_x[pa], lane, c, IN_DIM);
      v16h a1 = load_A_lds(sh_h0[qa], lane, c, HID);
      acc  = __builtin_amdgcn_wmma_f32_16x16x32_f16(false, a0, false, B_ih0[c],
                                                    (short)0, acc, false, false);
      acc1 = __builtin_amdgcn_wmma_f32_16x16x32_f16(false, a1, false, B_ih1[c],
                                                    (short)0, acc1, false, false);
    }
#pragma unroll
    for (int c = 0; c < 4; ++c) {
      v16h a0 = load_A_lds(sh_h0[qa], lane, c, HID);
      acc  = __builtin_amdgcn_wmma_f32_16x16x32_f16(false, a0, false, B_hh0[c],
                                                    (short)0, acc, false, false);
      if (c >= 2) {
        v16h a1 = load_A_lds(sh_h0[qa], lane, c, HID);
        acc1 = __builtin_amdgcn_wmma_f32_16x16x32_f16(false, a1, false, B_ih1[c],
                                                      (short)0, acc1, false, false);
      }
      v16h a2 = load_A_lds(sh_h1[pa], lane, c, HID);
      acc1 = __builtin_amdgcn_wmma_f32_16x16x32_f16(false, a2, false, B_hh1[c],
                                                    (short)0, acc1, false, false);
    }

#pragma unroll
    for (int j = 0; j < 8; ++j) acc[j]  = fast_tanh(acc[j]);
#pragma unroll
    for (int j = 0; j < 8; ++j) acc1[j] = fast_tanh(acc1[j]);

#pragma unroll
    for (int r = 0; r < 8; ++r) {
      sh_h0[pa][(rowoff + r) * HID + mycol] = (_Float16)acc[r];
      sh_h1[qa][(rowoff + r) * HID + mycol] = (_Float16)acc1[r];
      yout[((size_t)(t - 1) * BATCHN + bbase + rowoff + r) * HID + mycol] = acc1[r];
    }
    if (t == SEQ - 1) {
#pragma unroll
      for (int r = 0; r < 8; ++r)
        hn[(size_t)(bbase + rowoff + r) * HID + mycol] = acc[r];  // h0 final
    }
  }

  // ---- epilogue: h1(SEQ-1) ----
  __syncthreads();
  {
    const int qa = (SEQ - 1) & 1;  // buffer of h0(SEQ-1) and... h1(SEQ-2) is in (SEQ-2)&1
    v8f acc1;
#pragma unroll
    for (int j = 0; j < 8; ++j) acc1[j] = b1s;
#pragma unroll
    for (int c = 0; c < 4; ++c) {
      v16h a = load_A_lds(sh_h0[qa], lane, c, HID);
      acc1 = __builtin_amdgcn_wmma_f32_16x16x32_f16(false, a, false, B_ih1[c],
                                                    (short)0, acc1, false, false);
    }
#pragma unroll
    for (int c = 0; c < 4; ++c) {
      v16h a = load_A_lds(sh_h1[(SEQ - 2) & 1], lane, c, HID);
      acc1 = __builtin_amdgcn_wmma_f32_16x16x32_f16(false, a, false, B_hh1[c],
                                                    (short)0, acc1, false, false);
    }
#pragma unroll
    for (int j = 0; j < 8; ++j) acc1[j] = fast_tanh(acc1[j]);
#pragma unroll
    for (int r = 0; r < 8; ++r) {
      yout[((size_t)(SEQ - 1) * BATCHN + bbase + rowoff + r) * HID + mycol] = acc1[r];
      hn[(size_t)(BATCHN + bbase + rowoff + r) * HID + mycol] = acc1[r];  // h1 final
    }
  }
}

extern "C" void kernel_launch(void* const* d_in, const int* in_sizes, int n_in,
                              void* d_out, int out_size, void* d_ws, size_t ws_size,
                              hipStream_t stream) {
  const float* x    = (const float*)d_in[0];
  const float* Wih0 = (const float*)d_in[1];
  const float* bih0 = (const float*)d_in[2];
  const float* Whh0 = (const float*)d_in[3];
  const float* bhh0 = (const float*)d_in[4];
  const float* Wih1 = (const float*)d_in[5];
  const float* bih1 = (const float*)d_in[6];
  const float* Whh1 = (const float*)d_in[7];
  const float* bhh1 = (const float*)d_in[8];

  rnn2_fused<<<BATCHN / 16, 256, 0, stream>>>(x, Wih0, bih0, Whh0, bhh0,
                                              Wih1, bih1, Whh1, bhh1,
                                              (float*)d_out);
}
